// DistanceWeightedSampling_15187004359158
// MI455X (gfx1250) — matrix operand order repaired and checked by reading, not compiled, not run on hardware
//
#include <hip/hip_runtime.h>
#include <hip/hip_bf16.h>

// ---------------------------------------------------------------------------
// DistanceWeightedSampling for MI455X (gfx1250, wave32, WMMA + async-to-LDS)
//   x: [8192, 512] f32
//   out = [a_idx(57344) | x_a(57344x512) | x_p | x_n | x(8192x512)]  (all f32)
// ---------------------------------------------------------------------------

typedef __attribute__((ext_vector_type(16))) __bf16 v16bf;
typedef __attribute__((ext_vector_type(8)))  __bf16 v8bf;
typedef __attribute__((ext_vector_type(8)))  float  v8f;

#define NROWS 8192
#define DIM   512
#define K1    7          // negatives per row (batch_k - 1)
#define RPB   32         // rows per block in gram kernel (2 waves x 16 rows)
#define LDA   520        // padded LDS stride (elements) -> bank-conflict-free b128
#define LDB   520
#define JTILES (NROWS / 16)

// --- deterministic per-(s,i,j) Gumbel noise (split-mix style hash) ----------
__device__ __forceinline__ float gumbel_noise(unsigned s, unsigned i, unsigned j) {
  unsigned h = s * 0x9E3779B9u ^ (i * 0x85EBCA77u + 0x27D4EB2Fu) ^ (j * 0xC2B2AE3Du);
  h ^= h >> 16; h *= 0x7FEB352Du;
  h ^= h >> 15; h *= 0x846CA68Bu;
  h ^= h >> 16;
  float u = (float)(h >> 8) * (1.0f / 16777216.0f) + 5.9604645e-08f;
  return -__logf(-__logf(u));
}

// assemble a 16-element bf16 fragment from two contiguous 8-element LDS runs
__device__ __forceinline__ v16bf load_frag(const __bf16* p, int gap) {
  v8bf lo = *(const v8bf*)(p);
  v8bf hi = *(const v8bf*)(p + gap);
  v16bf r;
#pragma unroll
  for (int t = 0; t < 8; ++t) { r[t] = lo[t]; r[t + 8] = hi[t]; }
  return r;
}

// ---- CDNA5 async copy: global -> LDS, 16 bytes per active lane -------------
// GVS mode: LDS address from a VGPR; global address = SGPR base + VGPR byte offset
__device__ __forceinline__ void async_copy_b128(unsigned lds_off, unsigned long long gbase,
                                                unsigned goff) {
  asm volatile("global_load_async_to_lds_b128 %0, %1, %2"
               :: "v"(lds_off), "v"(goff), "s"(gbase) : "memory");
}
__device__ __forceinline__ void wait_asynccnt0() {
  asm volatile("s_wait_asynccnt 0" ::: "memory");
}

// ---------------------------------------------------------------------------
// Kernel 1: row-wise L2 normalize -> bf16 xn (ws), and copy x -> out tail
// ---------------------------------------------------------------------------
__global__ __launch_bounds__(128)
void dws_normalize_kernel(const float* __restrict__ x,
                          __bf16* __restrict__ xn,
                          float* __restrict__ out_x) {
  const int row = blockIdx.x;
  const int tid = threadIdx.x;           // 0..127
  const float4 v = ((const float4*)(x + (size_t)row * DIM))[tid];
  float ss = v.x * v.x + v.y * v.y + v.z * v.z + v.w * v.w;
#pragma unroll
  for (int off = 16; off > 0; off >>= 1) ss += __shfl_xor(ss, off, 32);
  __shared__ float partial[4];
  if ((tid & 31) == 0) partial[tid >> 5] = ss;
  __syncthreads();
  const float tot = partial[0] + partial[1] + partial[2] + partial[3];
  const float inv = 1.0f / fmaxf(__builtin_sqrtf(tot), 1e-12f);

  __bf16* xr = xn + (size_t)row * DIM + (size_t)tid * 4;
  xr[0] = (__bf16)(v.x * inv);
  xr[1] = (__bf16)(v.y * inv);
  xr[2] = (__bf16)(v.z * inv);
  xr[3] = (__bf16)(v.w * inv);
  ((float4*)(out_x + (size_t)row * DIM))[tid] = v;   // x passthrough output
}

// ---------------------------------------------------------------------------
// Kernel 2: fused WMMA Gram + distance weights + streaming Gumbel argmax,
// with double-buffered async global->LDS staging of the B (column) tiles.
// grid: (NROWS/RPB, K1) blocks x 64 threads (2 waves). Wave w owns 16 rows.
// ---------------------------------------------------------------------------
__global__ __launch_bounds__(64)
void dws_gram_sample_kernel(const __bf16* __restrict__ xn,
                            int* __restrict__ nidx) {
  __shared__ __bf16 Alds[RPB * LDA];      // 32 rows x 512 (padded)
  __shared__ __bf16 Blds[2][16 * LDB];    // double-buffered 16 cols x 512

  const int tid  = threadIdx.x;
  const int lane = tid & 31;
  const int wave = tid >> 5;                 // 0..1
  const unsigned s = blockIdx.y;             // sample index 0..6
  const int rowBase = blockIdx.x * RPB;
  const unsigned long long gbase = (unsigned long long)(const void*)xn;

  // ---- stage A: 32 rows of xn, b128 chunks (8 bf16 = 16B), plain loads ----
  for (int c = tid; c < RPB * (DIM / 8); c += 64) {
    const int r  = c >> 6;                   // /64 chunks per row
    const int cc = c & 63;
    *(uint4*)&Alds[r * LDA + cc * 8] =
        *(const uint4*)(xn + ((size_t)(rowBase + r)) * DIM + (size_t)cc * 8);
  }

  // ---- async-stage B tile 0 into buffer 0 ----
  {
    for (int c = tid; c < 16 * (DIM / 8); c += 64) {
      const int r  = c >> 6;
      const int cc = c & 63;
      const unsigned ldsoff = (unsigned)(uintptr_t)&Blds[0][r * LDB + cc * 8];
      const unsigned goff   = (unsigned)(((0 + r) * DIM + cc * 8) * sizeof(__bf16));
      async_copy_b128(ldsoff, gbase, goff);
    }
  }
  wait_asynccnt0();
  __syncthreads();

  const int arowL = wave * 16 + (lane & 15);             // local A row this lane reads
  const int ibase = rowBase + wave * 16 + ((lane & 16) ? 8 : 0);

  float best[8];
  int   bidx[8];
#pragma unroll
  for (int r = 0; r < 8; ++r) { best[r] = -3.0e38f; bidx[r] = 0; }

  for (int jt = 0; jt < JTILES; ++jt) {
    const __bf16* Bcur = Blds[jt & 1];
    const int col0 = jt * 16;

    // fire-and-forget async stage of the NEXT B tile into the other buffer
    if (jt + 1 < JTILES) {
      __bf16* Bnext = Blds[(jt + 1) & 1];
      const int ncol0 = col0 + 16;
      for (int c = tid; c < 16 * (DIM / 8); c += 64) {
        const int r  = c >> 6;
        const int cc = c & 63;
        const unsigned ldsoff = (unsigned)(uintptr_t)&Bnext[r * LDB + cc * 8];
        const unsigned goff   = (unsigned)(((ncol0 + r) * DIM + cc * 8) * sizeof(__bf16));
        async_copy_b128(ldsoff, gbase, goff);
      }
    }

    // ---- 16x16 sim tile over K = 512 : 16 WMMAs ----
    v8f acc = {};
#pragma unroll
    for (int kk = 0; kk < 16; ++kk) {
      // A fragment (16x32 bf16): lanes 0-15 K {0..7,16..23}, lanes 16-31 K {8..15,24..31}
      const int ak0 = kk * 32 + ((lane & 16) ? 8 : 0);
      v16bf Af = load_frag(&Alds[arowL * LDA + ak0], 16);
      // B fragment (32x16 bf16): lane n = lane&15, lanes 0-15 K {0..7,16..23}, lanes 16-31 K {8..15,24..31}
      const int bk0 = kk * 32 + ((lane & 16) ? 8 : 0);
      v16bf Bf = load_frag(&Bcur[(lane & 15) * LDB + bk0], 16);
      acc = __builtin_amdgcn_wmma_f32_16x16x32_bf16(
          false, Af, false, Bf, (short)0, acc, false, false);
    }

    // ---- epilogue: logits + streaming Gumbel argmax ----
    const int j = col0 + (lane & 15);
#pragma unroll
    for (int r = 0; r < 8; ++r) {
      const int i = ibase + r;
      float sim  = fminf(acc[r], 1.0f);
      float d2   = 2.0f - 2.0f * sim + ((i == j) ? 1.0f : 0.0f);
      float dist = fmaxf(__builtin_sqrtf(d2), 0.5f);
      // log_w = (2-d)*log(dist) - ((d-3)/2)*log(max(1 - dist^2/4, 1e-8)), d = 512
      float logw = -510.0f * __logf(dist)
                   - 254.5f * __logf(fmaxf(1.0f - 0.25f * dist * dist, 1e-8f));
      bool valid = ((i >> 3) != (j >> 3)) && (dist < 1.4f);
      float cand = (valid ? logw : -1.0e30f) + gumbel_noise(s, (unsigned)i, (unsigned)j);
      if (cand > best[r]) { best[r] = cand; bidx[r] = j; }
    }

    // next buffer must have landed; then all waves rendezvous
    if (jt + 1 < JTILES) wait_asynccnt0();
    __syncthreads();
  }

  // ---- argmax reduction across the 16 lanes sharing each row ----
#pragma unroll
  for (int r = 0; r < 8; ++r) {
    float v = best[r];
    int   id = bidx[r];
#pragma unroll
    for (int off = 8; off > 0; off >>= 1) {
      float ov = __shfl_xor(v, off, 16);
      int   oid = __shfl_xor(id, off, 16);
      if (ov > v || (ov == v && oid < id)) { v = ov; id = oid; }
    }
    if ((lane & 15) == 0) {
      const int i = ibase + r;
      nidx[(size_t)s * NROWS + i] = id;
    }
  }
}

// ---------------------------------------------------------------------------
// Kernel 3: build a_idx / p_idx / n_idx and gather rows into the output.
// grid: 57344 blocks x 128 threads (one float4 per thread per gathered row)
// ---------------------------------------------------------------------------
__global__ __launch_bounds__(128)
void dws_gather_kernel(const float* __restrict__ x,
                       const int* __restrict__ nidx,
                       float* __restrict__ out) {
  const int p   = blockIdx.x;          // 0..57343  (pair index = i*7 + t)
  const int tid = threadIdx.x;
  const int i   = p / K1;
  const int t   = p - i * K1;
  const int pos = i & 7;
  const int base = i & ~7;
  const int off = t + ((t >= pos) ? 1 : 0);
  const int pi  = base + off;
  const int ni  = nidx[(size_t)t * NROWS + i];

  const size_t NP = (size_t)NROWS * K1;          // 57344
  float* out_a  = out;
  float* out_xa = out + NP;
  float* out_xp = out_xa + NP * DIM;
  float* out_xn = out_xp + NP * DIM;

  if (tid == 0) out_a[p] = (float)i;

  const float4 va = ((const float4*)(x + (size_t)i  * DIM))[tid];
  const float4 vp = ((const float4*)(x + (size_t)pi * DIM))[tid];
  const float4 vn = ((const float4*)(x + (size_t)ni * DIM))[tid];
  ((float4*)(out_xa + (size_t)p * DIM))[tid] = va;
  ((float4*)(out_xp + (size_t)p * DIM))[tid] = vp;
  ((float4*)(out_xn + (size_t)p * DIM))[tid] = vn;
}

// ---------------------------------------------------------------------------
extern "C" void kernel_launch(void* const* d_in, const int* in_sizes, int n_in,
                              void* d_out, int out_size, void* d_ws, size_t ws_size,
                              hipStream_t stream) {
  const float* x = (const float*)d_in[0];
  float* out = (float*)d_out;

  // workspace: [xn bf16 8192x512 = 8 MiB][n_idx int 7x8192 = 224 KiB]
  __bf16* xn = (__bf16*)d_ws;
  int* nidx = (int*)((char*)d_ws + (size_t)NROWS * DIM * sizeof(__bf16));

  const size_t NP = (size_t)NROWS * K1;
  float* out_x = out + NP + 3 * NP * (size_t)DIM;   // x passthrough slot

  dws_normalize_kernel<<<NROWS, 128, 0, stream>>>(x, xn, out_x);
  dws_gram_sample_kernel<<<dim3(NROWS / RPB, K1), 64, 0, stream>>>(xn, nidx);
  dws_gather_kernel<<<(int)NP, 128, 0, stream>>>(x, nidx, out);
}
